// MixtralSparseMoeBlock_16587163697425
// MI455X (gfx1250) — matrix operand order repaired and checked by reading, not compiled, not run on hardware
//
#include <hip/hip_runtime.h>
#include <hip/hip_bf16.h>
#include <stdint.h>

#define T_TOK 4096
#define DDIM  1024
#define FDIM  4096
#define NEXP  8

typedef __attribute__((ext_vector_type(16))) __bf16 v16bf;
typedef __attribute__((ext_vector_type(2)))  __bf16 v2bf;
typedef __attribute__((ext_vector_type(8)))  float  v8f;

union Frag16 { v16bf v; uint32_t u[8]; };

// pack two f32 -> bf16x2 dword using native cvt (v_cvt_pk_bf16_f32)
__device__ __forceinline__ uint32_t bf16pk(float lo, float hi) {
#if __has_builtin(__builtin_amdgcn_cvt_pk_bf16_f32)
  v2bf r = __builtin_amdgcn_cvt_pk_bf16_f32(lo, hi);
  return __builtin_bit_cast(uint32_t, r);
#else
  union { v2bf v; uint32_t u; } r;
  r.v[0] = (__bf16)lo; r.v[1] = (__bf16)hi;
  return r.u;
#endif
}
__device__ __forceinline__ unsigned short bf16one(float f) {
  union { __bf16 h; unsigned short s; } r;
  r.h = (__bf16)f;
  return r.s;
}

// ---------------------------------------------------------------- init ------
__global__ __launch_bounds__(256)
void moe_init_kernel(float* __restrict__ out, int* __restrict__ counts, int n) {
  int i = blockIdx.x * blockDim.x + threadIdx.x;
  if (i < NEXP) counts[i] = 0;
  for (int j = i; j < n; j += gridDim.x * blockDim.x) out[j] = 0.f;
}

// -------------------------------------------------------------- router ------
// one wave per token: 8 logits, top-2, renormalized softmax weights,
// compaction into per-expert token/weight lists via atomics.
__global__ __launch_bounds__(256)
void moe_router_kernel(const float* __restrict__ x, const float* __restrict__ gw,
                       int* __restrict__ counts, int* __restrict__ tlist,
                       float* __restrict__ wlist) {
  const int wv   = threadIdx.x >> 5;
  const int lane = threadIdx.x & 31;
  const int t    = blockIdx.x * 8 + wv;
  if (t >= T_TOK) return;
  float acc[NEXP];
#pragma unroll
  for (int e = 0; e < NEXP; ++e) acc[e] = 0.f;
  const float* xr = x + (size_t)t * DDIM;
  for (int d = lane; d < DDIM; d += 32) {
    const float xv = xr[d];
#pragma unroll
    for (int e = 0; e < NEXP; ++e) acc[e] += xv * gw[e * DDIM + d];
  }
#pragma unroll
  for (int e = 0; e < NEXP; ++e) {
#pragma unroll
    for (int off = 16; off > 0; off >>= 1) acc[e] += __shfl_xor(acc[e], off, 32);
  }
  if (lane == 0) {
    int e1 = 0;
#pragma unroll
    for (int e = 1; e < NEXP; ++e) if (acc[e] > acc[e1]) e1 = e;
    int e2 = (e1 == 0) ? 1 : 0;
#pragma unroll
    for (int e = 0; e < NEXP; ++e) if (e != e1 && acc[e] > acc[e2]) e2 = e;
    // renormalized top-2 softmax == sigmoid of logit difference
    const float wa = __builtin_amdgcn_rcpf(1.f + __expf(acc[e2] - acc[e1]));
    const float wb = 1.f - wa;
    int p = atomicAdd(&counts[e1], 1);
    tlist[e1 * T_TOK + p] = t;  wlist[e1 * T_TOK + p] = wa;
    p = atomicAdd(&counts[e2], 1);
    tlist[e2 * T_TOK + p] = t;  wlist[e2 * T_TOK + p] = wb;
  }
}

// -------------------------------------------------------- expert GEMMs ------
// block = 256 threads = 8 waves, handles 16 gathered tokens of expert e.
// Wave w owns output columns d in [w*128, w*128+128) (8 C tiles, v8f each).
// F-loop in chunks of 128: stage A computes h chunk (each wave 16 columns,
// 2 bf16 WMMA accumulators over K=D), stage B does out += h * w2^T (K=128).
__global__ __launch_bounds__(256)
void moe_expert_kernel(const float* __restrict__ x,
                       const float* __restrict__ w1,
                       const float* __restrict__ w3,
                       const float* __restrict__ w2,
                       const int*   __restrict__ counts,
                       const int*   __restrict__ tlist,
                       const float* __restrict__ wlist,
                       float* __restrict__ out) {
  const int e    = blockIdx.y;
  const int nt   = counts[e];
  const int tile = blockIdx.x;
  if (tile * 16 >= nt) return;

  // padded LDS (row strides offset by 4 dwords -> no 64-bank conflicts)
  __shared__ __align__(16) uint32_t       xs[16][DDIM / 2 + 4]; // x tile, bf16x2
  __shared__ __align__(16) unsigned short hs[16][128 + 8];      // h chunk, bf16
  __shared__ int   tok_s[16];
  __shared__ float wgt_s[16];

  const int tid = threadIdx.x;
  if (tid < 16) {
    const int idx = tile * 16 + tid;
    int tok; float wg;
    if (idx < nt) { tok = tlist[e * T_TOK + idx]; wg = wlist[e * T_TOK + idx]; }
    else          { tok = tlist[e * T_TOK + tile * 16]; wg = 0.f; }
    tok_s[tid] = tok;  wgt_s[tid] = wg;
  }
  __syncthreads();

  { // gather + convert x tile to bf16 in LDS: 16 threads/row, 64 floats each
    const int row = tid >> 4, seg = tid & 15;
    const float2* src = (const float2*)(x + (size_t)tok_s[row] * DDIM + seg * 64);
    uint32_t* dst = &xs[row][seg * 32];
#pragma unroll
    for (int j = 0; j < 32; ++j) { const float2 f = src[j]; dst[j] = bf16pk(f.x, f.y); }
  }
  __syncthreads();

  const int wv   = tid >> 5;
  const int lane = tid & 31;
  const int arow = lane & 15;        // M row (A) or N column (B/C)
  const int kh8  = (lane >> 4) * 8;  // A-fragment K-half offset
  const int kb16 = (lane >> 4) * 16; // unused: B fragments use kh8 like A

  const v8f vz = {0.f, 0.f, 0.f, 0.f, 0.f, 0.f, 0.f, 0.f};
  v8f acc[8];
#pragma unroll
  for (int n = 0; n < 8; ++n) acc[n] = vz;

  const size_t wb13 = (size_t)e * FDIM * DDIM;
  const size_t wb2  = (size_t)e * DDIM * FDIM;

  for (int f0 = 0; f0 < FDIM; f0 += 128) {
    // ---- stage A: h[:, f0 + wv*16 + 0..15] = silu(x w1^T) * (x w3^T) ----
    v8f a1 = vz, a3 = vz;
    const float* w1p = w1 + wb13 + (size_t)(f0 + wv * 16 + arow) * DDIM;
    const float* w3p = w3 + wb13 + (size_t)(f0 + wv * 16 + arow) * DDIM;
#pragma unroll 2
    for (int k0 = 0; k0 < DDIM; k0 += 32) {
      Frag16 a, b1f, b3f;
      {
        const uint32_t* xp = &xs[arow][(k0 + kh8) >> 1];
        a.u[0] = xp[0]; a.u[1] = xp[1]; a.u[2] = xp[2]; a.u[3] = xp[3];
        const uint32_t* xq = &xs[arow][(k0 + 16 + kh8) >> 1];
        a.u[4] = xq[0]; a.u[5] = xq[1]; a.u[6] = xq[2]; a.u[7] = xq[3];
      }
      const float4* p1 = (const float4*)(w1p + k0 + kh8);
      const float4* p3 = (const float4*)(w3p + k0 + kh8);
#pragma unroll
      for (int q = 0; q < 4; ++q) {
        const float4 v = p1[q < 2 ? q : q + 2];
        b1f.u[2 * q]     = bf16pk(v.x, v.y);
        b1f.u[2 * q + 1] = bf16pk(v.z, v.w);
        const float4 u = p3[q < 2 ? q : q + 2];
        b3f.u[2 * q]     = bf16pk(u.x, u.y);
        b3f.u[2 * q + 1] = bf16pk(u.z, u.w);
      }
      a1 = __builtin_amdgcn_wmma_f32_16x16x32_bf16(false, a.v, false, b1f.v,
                                                   (short)0, a1, false, false);
      a3 = __builtin_amdgcn_wmma_f32_16x16x32_bf16(false, a.v, false, b3f.v,
                                                   (short)0, a3, false, false);
    }
    // silu(a1) * a3 -> hs (C layout: elem i is row kh8+i, col arow)
#pragma unroll
    for (int i = 0; i < 8; ++i) {
      const float v1 = a1[i];
      const float hv = v1 * __builtin_amdgcn_rcpf(1.f + __expf(-v1)) * a3[i];
      hs[kh8 + i][wv * 16 + arow] = bf16one(hv);
    }
    __syncthreads();

    // ---- stage B: acc += h_chunk @ w2[:, f0:f0+128]^T ----
    for (int kk = 0; kk < 128; kk += 32) {
      Frag16 ah;
      const uint32_t* hp = (const uint32_t*)&hs[arow][kk + kh8];
      ah.u[0] = hp[0]; ah.u[1] = hp[1]; ah.u[2] = hp[2]; ah.u[3] = hp[3];
      const uint32_t* hq = (const uint32_t*)&hs[arow][kk + 16 + kh8];
      ah.u[4] = hq[0]; ah.u[5] = hq[1]; ah.u[6] = hq[2]; ah.u[7] = hq[3];
      const int fk = f0 + kk + kh8;
#pragma unroll
      for (int n = 0; n < 8; ++n) {
        const int d = wv * 128 + n * 16 + arow;
        const float4* p2 = (const float4*)(w2 + wb2 + (size_t)d * FDIM + fk);
        Frag16 b2;
#pragma unroll
        for (int q = 0; q < 4; ++q) {
          const float4 v = p2[q < 2 ? q : q + 2];
          b2.u[2 * q]     = bf16pk(v.x, v.y);
          b2.u[2 * q + 1] = bf16pk(v.z, v.w);
        }
        acc[n] = __builtin_amdgcn_wmma_f32_16x16x32_bf16(false, ah.v, false, b2.v,
                                                         (short)0, acc[n], false, false);
      }
    }
    __syncthreads();  // protect hs before next chunk's writes
  }

  // ---- epilogue: out[tok, d] += acc * routing_weight (f32 atomics) ----
#pragma unroll
  for (int n = 0; n < 8; ++n) {
    const int d = wv * 128 + n * 16 + arow;
#pragma unroll
    for (int i = 0; i < 8; ++i) {
      const int m = kh8 + i;
      const float val = acc[n][i] * wgt_s[m];
      __hip_atomic_fetch_add(&out[(size_t)tok_s[m] * DDIM + d], val,
                             __ATOMIC_RELAXED, __HIP_MEMORY_SCOPE_AGENT);
    }
  }
}

// ---------------------------------------------------------------------------
extern "C" void kernel_launch(void* const* d_in, const int* in_sizes, int n_in,
                              void* d_out, int out_size, void* d_ws, size_t ws_size,
                              hipStream_t stream) {
  (void)in_sizes; (void)n_in; (void)out_size; (void)ws_size;
  const float* x  = (const float*)d_in[0];   // [T, D]
  const float* gw = (const float*)d_in[1];   // [E, D]
  const float* w1 = (const float*)d_in[2];   // [E, F, D]
  const float* w3 = (const float*)d_in[3];   // [E, F, D]
  const float* w2 = (const float*)d_in[4];   // [E, D, F]
  float* out = (float*)d_out;                // [T, D]

  char* ws = (char*)d_ws;
  int*   counts = (int*)ws;                                   // E ints
  int*   tlist  = (int*)(ws + 128);                           // E*T ints
  float* wlist  = (float*)(ws + 128 + (size_t)NEXP * T_TOK * 4);

  moe_init_kernel<<<1024, 256, 0, stream>>>(out, counts, T_TOK * DDIM);
  moe_router_kernel<<<T_TOK / 8, 256, 0, stream>>>(x, gw, counts, tlist, wlist);
  dim3 grid(T_TOK / 16, NEXP);   // expert-major dispatch -> per-expert L2 reuse
  moe_expert_kernel<<<grid, 256, 0, stream>>>(x, w1, w3, w2, counts, tlist, wlist, out);
}
